// MultiHeadAttentionLayer_80942953660861
// MI455X (gfx1250) — matrix operand order repaired and checked by reading, not compiled, not run on hardware
//
#include <hip/hip_runtime.h>

// ---------------------------------------------------------------------------
// MultiHeadAttention graph layer for MI455X (gfx1250, wave32).
//   Phase 1: fused GEMM  KQV[N][384] = h[N][128] @ [q | p | Wv]  (f16 WMMA,
//            fp32 accumulate, v_wmma_f32_16x16x32_f16)
//   Phase 2: per-edge scores + atomic scatter (L2-resident: K/Q/V = 76.8 MB
//            fits in the 192 MB L2, so random gathers/atomics stay on-chip)
//   Phase 3: normalize by z.
// ---------------------------------------------------------------------------

typedef __attribute__((ext_vector_type(16))) _Float16 v16h;
typedef __attribute__((ext_vector_type(8)))  _Float16 v8h;
typedef __attribute__((ext_vector_type(8)))  float    v8f;

#define IN_DIM 128
#define WCOLS  384          // K(128) | Q(128) | V(128)
#define LDSW   136          // LDS column stride in halfs: 272B = 16B-aligned

// ---------------------------------------------------------------------------
// Zero the accumulators (d_out and z) — must run every call.
// ---------------------------------------------------------------------------
__global__ __launch_bounds__(256) void zero_acc(float* __restrict__ out,
                                                float* __restrict__ z,
                                                int nNodes)
{
    const int i = blockIdx.x * 256 + threadIdx.x;
    const int tot = nNodes * 128;
    if (i < tot)        out[i] = 0.0f;
    if (i < nNodes * 8) z[i]   = 0.0f;
}

// ---------------------------------------------------------------------------
// KQV GEMM. Block = 256 threads = 8 waves; each wave owns a 16-row tile.
// A fragment (16x32 f16): lane L (l=L&15 -> M=l, hi=L>>4), elem j ->
//   K = hi*16 + j   (contiguous, same order as B)
// B fragment (32x16 f16): lane L (n=L&15), elem j -> K = (L>>4)*16 + j
// C/D (16x16 f32): lane L col = L&15, VGPR r -> row = r + 8*(L>>4)
// ---------------------------------------------------------------------------
__global__ __launch_bounds__(256) void kqv_gemm(
    const float* __restrict__ h,
    const float* __restrict__ q,     // -> cols   0..127  (K = h@q)
    const float* __restrict__ p,     // -> cols 128..255  (Q = h@p)
    const float* __restrict__ wv,    // -> cols 256..383  (V = h@Wv)
    float* __restrict__ kqv,
    int nNodes)
{
    __shared__ __align__(16) _Float16 Wlds[128 * LDSW]; // ~34 KB, col-major

    const int tid  = threadIdx.x;
    const int wave = tid >> 5;
    const int lane = tid & 31;
    const int l16  = lane & 15;
    const int hi   = lane >> 4;

    // m0 is wave-uniform: pin it to an SGPR so tile-full tests and the
    // store base stay scalar (no exec-mask churn in the common path).
    const int m0 = __builtin_amdgcn_readfirstlane(blockIdx.x * 128 + wave * 16);
    const bool fullTile = (m0 + 16 <= nNodes);

    int row = m0 + l16;
    if (row >= nNodes) row = nNodes - 1;          // clamp (WMMA needs full EXEC)
    const float* hrow = h + (long)row * IN_DIM;

    // Build all 4 K-step A fragments once; reused for all 24 output tiles.
    v16h a[4];
#pragma unroll
    for (int ks = 0; ks < 4; ++ks) {
#pragma unroll
        for (int jb = 0; jb < 2; ++jb) {
            const int base = ks * 32 + hi * 16 + jb * 8;
            const float4 f0 = *(const float4*)(hrow + base);
            const float4 f1 = *(const float4*)(hrow + base + 4);
            a[ks][jb * 8 + 0] = (_Float16)f0.x;
            a[ks][jb * 8 + 1] = (_Float16)f0.y;
            a[ks][jb * 8 + 2] = (_Float16)f0.z;
            a[ks][jb * 8 + 3] = (_Float16)f0.w;
            a[ks][jb * 8 + 4] = (_Float16)f1.x;
            a[ks][jb * 8 + 5] = (_Float16)f1.y;
            a[ks][jb * 8 + 6] = (_Float16)f1.z;
            a[ks][jb * 8 + 7] = (_Float16)f1.w;
        }
    }

    const float* const Wsrc[3] = { q, p, wv };

#pragma unroll
    for (int g = 0; g < 3; ++g) {
        __syncthreads();
        // Stage weight group into LDS: col-major f16, coalesced global reads.
        const float* wg = Wsrc[g];
        for (int idx = tid; idx < 128 * 128; idx += 256) {
            const int c = idx & 127;       // consecutive tid -> consecutive col
            const int k = idx >> 7;
            Wlds[c * LDSW + k] = (_Float16)wg[k * 128 + c];
        }
        __syncthreads();

#pragma unroll
        for (int ct = 0; ct < 8; ++ct) {
            const int col = ct * 16 + l16;
            v8f acc = {};
#pragma unroll
            for (int ks = 0; ks < 4; ++ks) {
                // 16 contiguous halfs: K = ks*32 + hi*16 .. +15  (16B aligned)
                const _Float16* bp = &Wlds[col * LDSW + ks * 32 + hi * 16];
                const v8h b0 = *(const v8h*)(bp);
                const v8h b1 = *(const v8h*)(bp + 8);
                const v16h b = __builtin_shufflevector(
                    b0, b1, 0,1,2,3,4,5,6,7,8,9,10,11,12,13,14,15);
                acc = __builtin_amdgcn_wmma_f32_16x16x32_f16(
                    false, a[ks], false, b, (short)0, acc, false, false);
            }

            // Store: lanes 0-15 write row m0+r (cols ct*16..+15, contiguous
            // 64B), lanes 16-31 write row m0+8+r. One base address + eight
            // immediate-offset global_store_b32 in the full-tile fast path.
            float* outp = kqv + (long)(m0 + 8 * hi) * WCOLS
                              + g * 128 + ct * 16 + l16;
            if (fullTile) {                       // scalar branch (SGPR cond)
#pragma unroll
                for (int r = 0; r < 8; ++r) outp[r * WCOLS] = acc[r];
            } else {
#pragma unroll
                for (int r = 0; r < 8; ++r)
                    if (m0 + 8 * hi + r < nNodes) outp[r * WCOLS] = acc[r];
            }
        }
    }
}

// ---------------------------------------------------------------------------
// Edge kernel: one wave per edge. lane = head(8) x quad(4); each lane owns 4
// of the 16 rank/out dims of its head. Dot-product reduced across the quad
// with shfl_xor; scores exp(clip(x/4,-5,5)); atomic scatter into d_out + z.
// e / src[e] / dst[e] are forced scalar (s_load + scalar guard).
// ---------------------------------------------------------------------------
__global__ __launch_bounds__(256) void edge_scatter(
    const int* __restrict__ src, const int* __restrict__ dst,
    const float* __restrict__ kqv,
    float* __restrict__ wv_out, float* __restrict__ z,
    int nEdges)
{
    const int e = __builtin_amdgcn_readfirstlane(blockIdx.x * 8
                                                 + (threadIdx.x >> 5));
    if (e >= nEdges) return;                      // scalar branch
    const int lane = threadIdx.x & 31;
    const int head = lane >> 2;
    const int quad = lane & 3;

    const int s = __builtin_amdgcn_readfirstlane(src[e]);   // s_load path
    const int d = __builtin_amdgcn_readfirstlane(dst[e]);

    const float4 kk = *(const float4*)(kqv + (long)s * WCOLS +       head * 16 + quad * 4);
    const float4 qq = *(const float4*)(kqv + (long)d * WCOLS + 128 + head * 16 + quad * 4);

    float sc = kk.x * qq.x + kk.y * qq.y + kk.z * qq.z + kk.w * qq.w;
    sc += __shfl_xor(sc, 1, 32);
    sc += __shfl_xor(sc, 2, 32);          // full 16-elem dot within the quad
    sc = __expf(fminf(fmaxf(sc * 0.25f, -5.0f), 5.0f));   // SCALE = sqrt(16)

    const float4 vv = *(const float4*)(kqv + (long)s * WCOLS + 256 + head * 16 + quad * 4);
    float* op = wv_out + (long)d * 128 + head * 16 + quad * 4;
    atomicAdd(op + 0, vv.x * sc);
    atomicAdd(op + 1, vv.y * sc);
    atomicAdd(op + 2, vv.z * sc);
    atomicAdd(op + 3, vv.w * sc);
    if (quad == 0) atomicAdd(z + (long)d * 8 + head, sc);
}

// ---------------------------------------------------------------------------
// out[n][h][d] /= (z[n][h] == 0 ? 0.001 : z[n][h])
// ---------------------------------------------------------------------------
__global__ __launch_bounds__(256) void normalize(float* __restrict__ out,
                                                 const float* __restrict__ z,
                                                 int nNodes)
{
    const int i = blockIdx.x * 256 + threadIdx.x;
    if (i >= nNodes * 128) return;
    const int n  = i >> 7;
    const int hd = (i >> 4) & 7;
    float zz = z[n * 8 + hd];
    zz = (zz == 0.0f) ? 0.001f : zz;
    out[i] = out[i] / zz;
}

// ---------------------------------------------------------------------------
extern "C" void kernel_launch(void* const* d_in, const int* in_sizes, int n_in,
                              void* d_out, int out_size, void* d_ws, size_t ws_size,
                              hipStream_t stream)
{
    const float* h   = (const float*)d_in[0];
    const int*   src = (const int*)  d_in[1];
    const int*   dst = (const int*)  d_in[2];
    const float* p   = (const float*)d_in[3];
    const float* q   = (const float*)d_in[4];
    const float* wvw = (const float*)d_in[5];

    const int nNodes = in_sizes[0] / IN_DIM;
    const int nEdges = in_sizes[1];

    float* kqv = (float*)d_ws;                          // [N][384]
    float* z   = kqv + (size_t)nNodes * WCOLS;          // [N][8]
    float* out = (float*)d_out;                         // [N][8][16]

    const int tot = nNodes * 128;
    zero_acc    <<<(tot + 255) / 256,    256, 0, stream>>>(out, z, nNodes);
    kqv_gemm    <<<(nNodes + 127) / 128, 256, 0, stream>>>(h, q, p, wvw, kqv, nNodes);
    edge_scatter<<<(nEdges + 7) / 8,     256, 0, stream>>>(src, dst, kqv, out, z, nEdges);
    normalize   <<<(tot + 255) / 256,    256, 0, stream>>>(out, z, nNodes);
}
